// SwinTransformerBlockAdapter_65721589563881
// MI455X (gfx1250) — matrix operand order repaired and checked by reading, not compiled, not run on hardware
//
#include <hip/hip_runtime.h>
#include <math.h>

// ---------------- constants ----------------
#define MROWS 25088      // B*L = 8*3136 tokens
#define CDIM  384
#define HID_  1536
#define NHEAD 12
#define DHEAD 32
#define NWIN  512        // B * 64 windows
#define NTOK  49         // 7*7
#define HW    56

typedef __attribute__((ext_vector_type(16))) _Float16 v16h;
typedef __attribute__((ext_vector_type(8)))  float    v8f;
typedef __attribute__((ext_vector_type(4)))  float    f4;
typedef __attribute__((ext_vector_type(4)))  _Float16 h4;

__device__ __forceinline__ v8f v8f_zero() {
  v8f z = {0.f,0.f,0.f,0.f,0.f,0.f,0.f,0.f};
  return z;
}

__device__ __forceinline__ v8f wmma_f32_f16(v16h a, v16h b, v8f c) {
  // (neg_a, A, neg_b, B, c_mod, C, reuse_a, reuse_b)
  return __builtin_amdgcn_wmma_f32_16x16x32_f16(false, a, false, b, (short)0, c, false, false);
}

// A-matrix (16x32 f16) fragment from LDS; per ISA: VGPR i holds
// k = (i<4?0:16) + half*8 + (i&3)*2 (+0/+1), row = lane&15.
__device__ __forceinline__ v16h ld_afrag(const _Float16* tile, int lane, int rstride) {
  int row = lane & 15, half = lane >> 4;
  const _Float16* p = tile + row * rstride;
  v16h a;
#pragma unroll
  for (int i = 0; i < 8; ++i) {
    int k = ((i >> 2) << 4) + half * 8 + ((i & 3) << 1);
    a[2*i]   = p[k];
    a[2*i+1] = p[k+1];
  }
  return a;
}

// B-matrix (32x16 f16) fragment: col = lane&15, k = (i<4?0:16) + half*8 + (i&3)*2 (+0/+1).
// Element (k,col) read at tile[k*kstep + col*cstep].
__device__ __forceinline__ v16h ld_bfrag(const _Float16* tile, int lane, int kstep, int cstep) {
  int col = lane & 15, half = lane >> 4;
  v16h b;
#pragma unroll
  for (int i = 0; i < 8; ++i) {
    int k = ((i >> 2) << 4) + half * 8 + ((i & 3) << 1);
    b[2*i]   = tile[k*kstep       + col*cstep];
    b[2*i+1] = tile[(k+1)*kstep   + col*cstep];
  }
  return b;
}

__device__ __forceinline__ float gelu_exact(float x) {
  return 0.5f * x * (1.0f + erff(x * 0.70710678118654752f));
}

// ---------------- generic WMMA GEMM ----------------
// out[M,N] = act(A[M,K] @ W[K,N] + bias) (+ res[M,N]); fp32 memory, f16 WMMA.
// FULL: M%64==0 && K%32==0 && N%64==0 -> branch-free vectorized staging.
#define TM 64
#define TN 64
#define TK 32

template <bool FULL>
__global__ __launch_bounds__(128)
void gemm_kernel(const float* __restrict__ A, const float* __restrict__ W,
                 const float* __restrict__ bias, const float* __restrict__ res,
                 float* __restrict__ out, int M, int K, int N, int act)
{
  __shared__ _Float16 As[TM * TK];       // [row][k]
  __shared__ _Float16 Bs[TN * TK];       // [col][k]  (transposed stage)
  const int tid  = threadIdx.x;
  const int wave = tid >> 5, lane = tid & 31;
  const int tMr = blockIdx.y * TM, tNc = blockIdx.x * TN;

  v8f acc[4] = { v8f_zero(), v8f_zero(), v8f_zero(), v8f_zero() };

  for (int k0 = 0; k0 < K; k0 += TK) {
    if constexpr (FULL) {
      // ---- A tile: 64x32 fp32 -> f16, 4 x b128 loads + 4 x ds_store_b64 per thread
#pragma unroll
      for (int it = 0; it < 4; ++it) {
        int t  = tid + it * 128;            // 0..511 float4 chunks
        int r  = t >> 3;                    // 8 chunks per row
        int c4 = (t & 7) << 2;
        f4 v = *reinterpret_cast<const f4*>(A + (size_t)(tMr + r) * K + (k0 + c4));
        h4 h = { (_Float16)v[0], (_Float16)v[1], (_Float16)v[2], (_Float16)v[3] };
        *reinterpret_cast<h4*>(As + r * TK + c4) = h;
      }
      // ---- B tile: 32x64 fp32 -> f16 transposed to [col][k]; one 4x4 block/thread
      {
        int kb = (tid >> 4) << 2;           // 0,4,...,28
        int cb = (tid & 15) << 2;           // 0,4,...,60
        const float* wp = W + (size_t)(k0 + kb) * N + (tNc + cb);
        f4 r0 = *reinterpret_cast<const f4*>(wp);
        f4 r1 = *reinterpret_cast<const f4*>(wp + (size_t)N);
        f4 r2 = *reinterpret_cast<const f4*>(wp + 2 * (size_t)N);
        f4 r3 = *reinterpret_cast<const f4*>(wp + 3 * (size_t)N);
        h4 c0 = { (_Float16)r0[0], (_Float16)r1[0], (_Float16)r2[0], (_Float16)r3[0] };
        h4 c1 = { (_Float16)r0[1], (_Float16)r1[1], (_Float16)r2[1], (_Float16)r3[1] };
        h4 c2 = { (_Float16)r0[2], (_Float16)r1[2], (_Float16)r2[2], (_Float16)r3[2] };
        h4 c3 = { (_Float16)r0[3], (_Float16)r1[3], (_Float16)r2[3], (_Float16)r3[3] };
        *reinterpret_cast<h4*>(Bs + (cb + 0) * TK + kb) = c0;
        *reinterpret_cast<h4*>(Bs + (cb + 1) * TK + kb) = c1;
        *reinterpret_cast<h4*>(Bs + (cb + 2) * TK + kb) = c2;
        *reinterpret_cast<h4*>(Bs + (cb + 3) * TK + kb) = c3;
      }
      // speculative prefetch of next K-tile (global_prefetch_b8)
      if (k0 + TK < K) {
        __builtin_prefetch(A + (size_t)(tMr + (tid >> 1)) * K + (k0 + TK), 0, 0);
        __builtin_prefetch(W + (size_t)(k0 + TK + (tid & 31)) * N + tNc, 0, 0);
      }
    } else {
      // ---- guarded slow path (edge tiles / unaligned K,N)
      for (int t = tid; t < TM * TK; t += 128) {
        int r = t >> 5, c = t & 31;
        int gr = tMr + r, gc = k0 + c;
        float v = (gr < M && gc < K) ? A[(size_t)gr * K + gc] : 0.f;
        As[t] = (_Float16)v;
      }
      for (int t = tid; t < TN * TK; t += 128) {
        int n = t >> 5, c = t & 31;
        int gk = k0 + c, gn = tNc + n;
        float v = (gk < K && gn < N) ? W[(size_t)gk * N + gn] : 0.f;
        Bs[n * TK + c] = (_Float16)v;
      }
    }
    __syncthreads();

    v16h af = ld_afrag(As + wave * 16 * TK, lane, TK);
#pragma unroll
    for (int sub = 0; sub < 4; ++sub) {
      v16h bf = ld_bfrag(Bs + sub * 16 * TK, lane, /*kstep*/1, /*cstep*/TK);
      acc[sub] = wmma_f32_f16(af, bf, acc[sub]);
    }
    __syncthreads();
  }

  // epilogue: C/D layout: lane col = lane&15; VGPR r -> row r + (lane>>4)*8
  const int half = lane >> 4, nl = lane & 15;
  const int row0 = tMr + wave * 16;
#pragma unroll
  for (int sub = 0; sub < 4; ++sub) {
    int col = tNc + sub * 16 + nl;
    if (!FULL && col >= N) continue;
    float bv = bias ? bias[col] : 0.f;
#pragma unroll
    for (int r = 0; r < 8; ++r) {
      int grow = row0 + half * 8 + r;
      if (!FULL && grow >= M) continue;
      float v = acc[sub][r] + bv;
      if (act == 1) v = gelu_exact(v);
      if (res) v += res[(size_t)grow * N + col];
      out[(size_t)grow * N + col] = v;
    }
  }
}

// ---------------- relative position bias table ----------------
__global__ void bias_table_kernel(const float* __restrict__ rpb, float* __restrict__ tab)
{
  int head = blockIdx.x;
  for (int t = threadIdx.x; t < NTOK * NTOK; t += blockDim.x) {
    int i = t / NTOK, j = t % NTOK;
    int dy = (i / 7) - (j / 7) + 6;
    int dx = (i % 7) - (j % 7) + 6;
    tab[head * NTOK * NTOK + t] = rpb[(dy * 13 + dx) * NHEAD + head];
  }
}

// ---------------- LN (+ optional shift & window partition) ----------------
// windowed==1: dest row d is window-partition order of the (rolled by -shift) image.
__global__ __launch_bounds__(128)
void ln_part_kernel(const float* __restrict__ in, const float* __restrict__ g,
                    const float* __restrict__ b, float* __restrict__ out,
                    int shift, int windowed)
{
  int wave = threadIdx.x >> 5, lane = threadIdx.x & 31;
  int d = blockIdx.x * 4 + wave;
  if (d >= MROWS) return;
  int src = d;
  if (windowed) {
    int gw = d / NTOK, inner = d % NTOK;
    int bb = gw >> 6, wi = gw & 63;
    int h = (wi >> 3) * 7 + inner / 7;
    int w = (wi & 7) * 7 + inner % 7;
    int hs = (h + shift) % HW, ws = (w + shift) % HW;
    src = bb * (HW * HW) + hs * HW + ws;
  }
  const float* row = in + (size_t)src * CDIM;
  f4 v[3];
  float s = 0.f, sq = 0.f;
#pragma unroll
  for (int i = 0; i < 3; ++i) {
    v[i] = *reinterpret_cast<const f4*>(row + (lane + i * 32) * 4);
#pragma unroll
    for (int e = 0; e < 4; ++e) { s += v[i][e]; sq += v[i][e] * v[i][e]; }
  }
#pragma unroll
  for (int off = 16; off > 0; off >>= 1) {
    s  += __shfl_xor(s,  off, 32);
    sq += __shfl_xor(sq, off, 32);
  }
  float mean = s / (float)CDIM;
  float var  = sq / (float)CDIM - mean * mean;
  float inv  = rsqrtf(var + 1e-5f);
  float* orow = out + (size_t)d * CDIM;
#pragma unroll
  for (int i = 0; i < 3; ++i) {
    int c = (lane + i * 32) * 4;
    f4 gg = *reinterpret_cast<const f4*>(g + c);
    f4 bb2 = *reinterpret_cast<const f4*>(b + c);
    f4 o;
#pragma unroll
    for (int e = 0; e < 4; ++e) o[e] = (v[i][e] - mean) * inv * gg[e] + bb2[e];
    *reinterpret_cast<f4*>(orow + c) = o;
  }
}

// ---------------- residual + window-reverse + roll(+3,+3) gather ----------------
__global__ void addroll_kernel(const float* __restrict__ base,
                               const float* __restrict__ aw,
                               float* __restrict__ out)
{
  size_t i4 = (size_t)blockIdx.x * 256 + threadIdx.x;     // float4 index
  if (i4 >= (size_t)MROWS * (CDIM / 4)) return;
  int c4 = (int)(i4 % (CDIM / 4)) * 4;
  int token = (int)(i4 / (CDIM / 4));
  int bb = token / (HW * HW), rem = token % (HW * HW);
  int h = rem / HW, w = rem % HW;
  int hp = (h + HW - 3) % HW, wp = (w + HW - 3) % HW;     // un-roll by +3
  int wi = (hp / 7) * 8 + (wp / 7);
  int inner = (hp % 7) * 7 + (wp % 7);
  size_t srow = (size_t)(bb * 64 + wi) * NTOK + inner;
  f4 a = *reinterpret_cast<const f4*>(base + (size_t)token * CDIM + c4);
  f4 g = *reinterpret_cast<const f4*>(aw + srow * CDIM + c4);
  f4 o;
#pragma unroll
  for (int e = 0; e < 4; ++e) o[e] = a[e] + g[e];
  *reinterpret_cast<f4*>(out + (size_t)token * CDIM + c4) = o;
}

// ---------------- windowed attention (one block per window*head) ----------------
__global__ __launch_bounds__(128)
void attn_kernel(const float* __restrict__ qkv, const float* __restrict__ biasTab,
                 const int* __restrict__ maskFlag, float* __restrict__ out)
{
  __shared__ _Float16 Qs[64 * 32];
  __shared__ _Float16 Ks[64 * 32];
  __shared__ _Float16 Vs[64 * 32];
  __shared__ float    Ss[64 * 64];
  __shared__ _Float16 Ps[64 * 64];

  const int blk  = blockIdx.x;
  const int win  = blk / NHEAD, head = blk % NHEAD;
  const int tid  = threadIdx.x, wave = tid >> 5, lane = tid & 31;
  const float scale = 0.17677669529663687f;   // 32^-0.5

  // stage Q (pre-scaled), K, V as f16; pad rows 49..63 with 0
  const size_t base = (size_t)win * NTOK * (3 * CDIM) + (size_t)head * DHEAD;
#pragma unroll
  for (int it = 0; it < 4; ++it) {
    int t  = tid + it * 128;                 // 0..511 float4 chunks
    int n  = t >> 3, c4 = (t & 7) << 2;
    h4 hq = { (_Float16)0.f, (_Float16)0.f, (_Float16)0.f, (_Float16)0.f };
    h4 hk = hq, hv = hq;
    if (n < NTOK) {
      const float* rp = qkv + base + (size_t)n * (3 * CDIM) + c4;
      f4 q = *reinterpret_cast<const f4*>(rp);
      f4 k = *reinterpret_cast<const f4*>(rp + CDIM);
      f4 v = *reinterpret_cast<const f4*>(rp + 2 * CDIM);
#pragma unroll
      for (int e = 0; e < 4; ++e) {
        hq[e] = (_Float16)(q[e] * scale);
        hk[e] = (_Float16)k[e];
        hv[e] = (_Float16)v[e];
      }
    }
    *reinterpret_cast<h4*>(Qs + n * 32 + c4) = hq;
    *reinterpret_cast<h4*>(Ks + n * 32 + c4) = hk;
    *reinterpret_cast<h4*>(Vs + n * 32 + c4) = hv;
  }
  __syncthreads();

  // S = Q @ K^T  (64x64, single K-step of 32)
  {
    v16h aq = ld_afrag(Qs + wave * 16 * 32, lane, 32);
    const int half = lane >> 4, nl = lane & 15;
#pragma unroll
    for (int sub = 0; sub < 4; ++sub) {
      // B(k,col) = K[col][k]  -> tile = Ks + sub*16*32, kstep=1, cstep=32
      v16h bk = ld_bfrag(Ks + sub * 16 * 32, lane, 1, 32);
      v8f s = wmma_f32_f16(aq, bk, v8f_zero());
#pragma unroll
      for (int r = 0; r < 8; ++r)
        Ss[(wave * 16 + half * 8 + r) * 64 + sub * 16 + nl] = s[r];
    }
  }
  __syncthreads();

  // softmax rows (bias + shift-mask), write P (f16, padded with zeros)
  const int useMask = maskFlag ? maskFlag[0] : 0;
  if (tid < 64) {
    int i = tid;
    if (i < NTOK) {
      int wimg = win & 63;
      int wh = wimg >> 3, ww = wimg & 7;
      int cnti = 0;
      if (useMask) {
        int hi = wh * 7 + i / 7, wi2 = ww * 7 + i % 7;
        int rh = hi < 49 ? 0 : (hi < 53 ? 1 : 2);
        int rw = wi2 < 49 ? 0 : (wi2 < 53 ? 1 : 2);
        cnti = rh * 3 + rw;
      }
      const float* bt = biasTab + head * NTOK * NTOK + i * NTOK;
      float mx = -1e30f;
      for (int j = 0; j < NTOK; ++j) {
        float v = Ss[i * 64 + j] + bt[j];
        if (useMask) {
          int hj = wh * 7 + j / 7, wj = ww * 7 + j % 7;
          int rh = hj < 49 ? 0 : (hj < 53 ? 1 : 2);
          int rw = wj < 49 ? 0 : (wj < 53 ? 1 : 2);
          if (rh * 3 + rw != cnti) v -= 100.f;
        }
        Ss[i * 64 + j] = v;
        mx = fmaxf(mx, v);
      }
      float sum = 0.f;
      for (int j = 0; j < NTOK; ++j) {
        float e = __expf(Ss[i * 64 + j] - mx);
        Ss[i * 64 + j] = e;
        sum += e;
      }
      float inv = 1.f / sum;
      for (int j = 0; j < NTOK; ++j) Ps[i * 64 + j] = (_Float16)(Ss[i * 64 + j] * inv);
      for (int j = NTOK; j < 64; ++j) Ps[i * 64 + j] = (_Float16)0.f;
    } else {
      for (int j = 0; j < 64; ++j) Ps[i * 64 + j] = (_Float16)0.f;
    }
  }
  __syncthreads();

  // O = P @ V  (64x32, two K-steps of 32)
  v8f oacc[2] = { v8f_zero(), v8f_zero() };
#pragma unroll
  for (int ks = 0; ks < 2; ++ks) {
    v16h ap = ld_afrag(Ps + wave * 16 * 64 + ks * 32, lane, 64);
#pragma unroll
    for (int sub = 0; sub < 2; ++sub) {
      // B(k,col) = V[ks*32 + k][sub*16 + col] -> kstep=32, cstep=1
      v16h bv = ld_bfrag(Vs + ks * 32 * 32 + sub * 16, lane, 32, 1);
      oacc[sub] = wmma_f32_f16(ap, bv, oacc[sub]);
    }
  }
  const int half = lane >> 4, nl = lane & 15;
#pragma unroll
  for (int sub = 0; sub < 2; ++sub)
#pragma unroll
    for (int r = 0; r < 8; ++r) {
      int row = wave * 16 + half * 8 + r;
      if (row < NTOK)
        out[((size_t)win * NTOK + row) * CDIM + head * DHEAD + sub * 16 + nl] = oacc[sub][r];
    }
}

// ---------------- host launcher ----------------
extern "C" void kernel_launch(void* const* d_in, const int* in_sizes, int n_in,
                              void* d_out, int out_size, void* d_ws, size_t ws_size,
                              hipStream_t stream)
{
  (void)in_sizes; (void)n_in; (void)out_size; (void)ws_size;
  const float* x_FMAG = (const float*)d_in[0];
  const float* x      = (const float*)d_in[1];
  const float* z      = (const float*)d_in[2];
  const int*   mask_z = (const int*)  d_in[3];
  const float* ln1_g  = (const float*)d_in[4];
  const float* ln1_b  = (const float*)d_in[5];
  const float* ln2_g  = (const float*)d_in[6];
  const float* ln2_b  = (const float*)d_in[7];
  const float* w_qkv  = (const float*)d_in[8];
  const float* b_qkv  = (const float*)d_in[9];
  const float* w_proj = (const float*)d_in[10];
  const float* b_proj = (const float*)d_in[11];
  const float* rpb    = (const float*)d_in[12];
  const float* w_fc1  = (const float*)d_in[13];
  const float* b_fc1  = (const float*)d_in[14];
  const float* w_fc2  = (const float*)d_in[15];
  const float* b_fc2  = (const float*)d_in[16];
  const float* st_dw  = (const float*)d_in[17];
  const float* st_db  = (const float*)d_in[18];
  const float* st_uw  = (const float*)d_in[19];
  const float* st_ub  = (const float*)d_in[20];
  const float* t_dw   = (const float*)d_in[21];
  const float* t_db   = (const float*)d_in[22];
  const float* t_uw   = (const float*)d_in[23];
  const float* t_ub   = (const float*)d_in[24];
  const float* t2_dw  = (const float*)d_in[25];
  const float* t2_db  = (const float*)d_in[26];
  const float* t2_uw  = (const float*)d_in[27];
  const float* t2_ub  = (const float*)d_in[28];

  float* out = (float*)d_out;
  const size_t M = MROWS, C = CDIM, MC = M * C;

  // workspace layout
  float* ws      = (float*)d_ws;
  float* biasTab = ws;                       // NH*49*49  (< 32768)
  float* smallb  = biasTab + 32768;          // M*64 (adapter bottlenecks)
  float* x0      = smallb + M * 64;          // shortcut_x
  float* xw      = x0 + MC;                  // windowed LN / LN scratch
  float* att     = xw + MC;                  // pre-proj attention out
  float* x1      = att + MC;
  float* z1      = x1 + MC;
  float* tmp     = z1 + MC;
  float* big     = tmp + MC;                 // M*1536 (qkv and MLP hidden)

  auto gemm = [&](const float* A, const float* W, const float* bias, const float* res,
                  float* o, int m, int k, int n, int act) {
    dim3 grid((n + TN - 1) / TN, (m + TM - 1) / TM);
    bool full = (m % TM == 0) && (k % TK == 0) && (n % TN == 0);
    if (full)
      gemm_kernel<true ><<<grid, 128, 0, stream>>>(A, W, bias, res, o, m, k, n, act);
    else
      gemm_kernel<false><<<grid, 128, 0, stream>>>(A, W, bias, res, o, m, k, n, act);
  };
  const int lnBlocks  = (MROWS + 3) / 4;
  const int ewBlocks  = (int)((M * (C / 4) + 255) / 256);

  // relative-position bias table
  bias_table_kernel<<<NHEAD, 256, 0, stream>>>(rpb, biasTab);

  // adapter_st: x0 = GELU(x_FMAG @ st_dw + st_db) @ st_uw + st_ub + x
  gemm(x_FMAG, st_dw, st_db, nullptr, smallb, M, C, 48, 1);
  gemm(smallb, st_uw, st_ub, x, x0, M, 48, C, 0);

  // ---- x branch: LN -> window partition (no shift) -> attn (no mask) -> proj -> awx
  ln_part_kernel<<<lnBlocks, 128, 0, stream>>>(x0, ln1_g, ln1_b, xw, 0, 1);
  gemm(xw, w_qkv, b_qkv, nullptr, big, M, C, 3 * C, 0);
  attn_kernel<<<NWIN * NHEAD, 128, 0, stream>>>(big, biasTab, nullptr, att);
  gemm(att, w_proj, b_proj, nullptr, out + MC, M, C, C, 0);         // awx4

  // ---- z branch: LN -> shift(-3) -> window partition -> attn (mask) -> proj -> awz
  ln_part_kernel<<<lnBlocks, 128, 0, stream>>>(z, ln1_g, ln1_b, xw, 3, 1);
  gemm(xw, w_qkv, b_qkv, nullptr, big, M, C, 3 * C, 0);
  attn_kernel<<<NWIN * NHEAD, 128, 0, stream>>>(big, biasTab, mask_z, att);
  gemm(att, w_proj, b_proj, nullptr, out + 3 * MC, M, C, C, 0);     // awz4

  // ---- x1 = shortcut_x + roll(wrev(awx)) + ad_t(shortcut_z)
  gemm(z, t_dw, t_db, nullptr, smallb, M, C, 16, 0);
  addroll_kernel<<<ewBlocks, 256, 0, stream>>>(x0, out + MC, tmp);
  gemm(smallb, t_uw, t_ub, tmp, x1, M, 16, C, 0);

  // ---- z1 = shortcut_z + roll(wrev(awz)) + ad_t(shortcut_x)
  gemm(x0, t_dw, t_db, nullptr, smallb, M, C, 16, 0);
  addroll_kernel<<<ewBlocks, 256, 0, stream>>>(z, out + 3 * MC, tmp);
  gemm(smallb, t_uw, t_ub, tmp, z1, M, 16, C, 0);

  // ---- x2 = x1 + mlp(LN2(x1)) + ad_t2(z1)
  ln_part_kernel<<<lnBlocks, 128, 0, stream>>>(x1, ln2_g, ln2_b, xw, 0, 0);
  gemm(xw, w_fc1, b_fc1, nullptr, big, M, C, HID_, 1);
  gemm(big, w_fc2, b_fc2, x1, tmp, M, HID_, C, 0);
  gemm(z1, t2_dw, t2_db, nullptr, smallb, M, C, 16, 0);
  gemm(smallb, t2_uw, t2_ub, tmp, out, M, 16, C, 0);                // x2

  // ---- z2 = z1 + mlp(LN2(z1)) + ad_t2(x1)
  ln_part_kernel<<<lnBlocks, 128, 0, stream>>>(z1, ln2_g, ln2_b, xw, 0, 0);
  gemm(xw, w_fc1, b_fc1, nullptr, big, M, C, HID_, 1);
  gemm(big, w_fc2, b_fc2, z1, tmp, M, HID_, C, 0);
  gemm(x1, t2_dw, t2_db, nullptr, smallb, M, C, 16, 0);
  gemm(smallb, t2_uw, t2_ub, tmp, out + 2 * MC, M, 16, C, 0);       // z2
}
